// Net_49538152792195
// MI455X (gfx1250) — matrix operand order repaired and checked by reading, not compiled, not run on hardware
//
#include <hip/hip_runtime.h>
#include <hip/hip_bf16.h>
#include <stdint.h>

typedef __attribute__((ext_vector_type(16))) _Float16 v16h;
typedef __attribute__((ext_vector_type(8)))  float    v8f;

#define BN_EPS 1e-5f

__device__ __forceinline__ float sigmoidf_(float x){ return 1.f/(1.f+__expf(-x)); }

// ---------------- generic zero ----------------
__global__ void k_zero_u32(uint32_t* __restrict__ p, int n){
  int i = blockIdx.x*blockDim.x + threadIdx.x;
  if (i < n) p[i] = 0u;
}

// -------- f32 -> f16 weight conversion, optional K zero-padding --------
__global__ void k_cvt_pad(const float* __restrict__ src, _Float16* __restrict__ dst,
                          int rows, int kin, int kout){
  int i = blockIdx.x*blockDim.x + threadIdx.x;
  if (i >= rows*kout) return;
  int r = i / kout, k = i - r*kout;
  dst[i] = (k < kin) ? (_Float16)src[(size_t)r*kin + k] : (_Float16)0.f;
}

// -------- bit-plane quant + three 1x1 LUT convs for one pixel --------
__device__ __forceinline__ void quant9(float xr, float xg, float xb,
                                       const float* __restrict__ wqs,
                                       const float* __restrict__ bqs,
                                       float out9[9]){
  int v[3]; v[0]=(int)rintf(xr); v[1]=(int)rintf(xg); v[2]=(int)rintf(xb);
  #pragma unroll
  for (int col=0; col<3; ++col){
    #pragma unroll
    for (int o=0; o<3; ++o){
      int ch = col*3 + o;
      float acc = bqs[ch];
      const float* w = wqs + ch*8;
      #pragma unroll
      for (int j=0; j<8; ++j)                       // MSB-first bit planes
        acc += w[j] * (float)((v[col] >> (7-j)) & 1);
      out9[ch] = acc;
    }
  }
}

__device__ __forceinline__ void load_wq(int t,
                                        const float* wr,const float* br,
                                        const float* wg,const float* bg,
                                        const float* wb,const float* bb,
                                        float* wqs, float* bqs){
  if (t < 72){
    int ch = t>>3, j = t&7, col = ch/3, o = ch%3;
    const float* w = (col==0)?wr:((col==1)?wg:wb);
    wqs[t] = w[o*8 + j];
  }
  if (t < 9){
    int col = t/3, o = t%3;
    const float* bsrc = (col==0)?br:((col==1)?bg:bb);
    bqs[t] = bsrc[o];
  }
}

// Stats layout in S (floats): sum[stage*64+c], sq[256+stage*64+c],
//                             scale[512+stage*64+c], bias[768+stage*64+c]
// stage0: 9ch x9 | stage1: 16ch z0 | stage2: 32ch z1 | stage3: 64ch z2

// ---------------- K1: batch stats of the 9-channel LUT output ----------------
__global__ void k_stats9(const float* __restrict__ x,
                         const float* wr,const float* br,const float* wg,const float* bg,
                         const float* wb,const float* bb,
                         float* __restrict__ S){
  __shared__ float wqs[72], bqs[9], ssum[9], ssq[9];
  int t = threadIdx.x, b = blockIdx.x;
  load_wq(t, wr,br,wg,bg,wb,bb, wqs,bqs);
  if (t < 9){ ssum[t]=0.f; ssq[t]=0.f; }
  __syncthreads();
  const float* xb_ = x + (size_t)b*3*1024;
  float ls[9], lq[9];
  #pragma unroll
  for (int c=0;c<9;++c){ ls[c]=0.f; lq[c]=0.f; }
  for (int p=t; p<1024; p+=256){
    float c9[9];
    quant9(xb_[p], xb_[1024+p], xb_[2048+p], wqs, bqs, c9);
    #pragma unroll
    for (int c=0;c<9;++c){ ls[c]+=c9[c]; lq[c]+=c9[c]*c9[c]; }
  }
  #pragma unroll
  for (int c=0;c<9;++c){ atomicAdd(&ssum[c], ls[c]); atomicAdd(&ssq[c], lq[c]); }
  __syncthreads();
  if (t < 9){ atomicAdd(&S[t], ssum[t]); atomicAdd(&S[256+t], ssq[t]); }
}

// ---------------- finalize BN: mean/var -> scale/bias ----------------
__global__ void k_finalize(float* __restrict__ S, int stage, int nch, float invCount){
  int t = threadIdx.x;
  if (t < nch){
    float m = S[stage*64+t]*invCount;
    float q = S[256+stage*64+t]*invCount;
    float sc = rsqrtf(q - m*m + BN_EPS);
    S[512+stage*64+t] = sc;
    S[768+stage*64+t] = -m*sc;
  }
}

// ---------------- K3: bn9+sigmoid -> conv0 (9->16, 3x3 s2 p1), stats16 ----------------
__global__ void __launch_bounds__(256)
k_conv0(const float* __restrict__ x,
        const float* wr,const float* br,const float* wg,const float* bg,
        const float* wb,const float* bb,
        const float* __restrict__ wc0, const float* __restrict__ bc0,
        float* __restrict__ S, float* __restrict__ z0){
  __shared__ float a9[9][1024];
  __shared__ float wcs[16*9*9];
  __shared__ float wqs[72], bqs[9], ssum[16], ssq[16];
  int t = threadIdx.x, b = blockIdx.x;
  load_wq(t, wr,br,wg,bg,wb,bb, wqs,bqs);
  if (t < 16){ ssum[t]=0.f; ssq[t]=0.f; }
  for (int i=t; i<1296; i+=256) wcs[i] = wc0[i];
  __syncthreads();
  const float* xb_ = x + (size_t)b*3072;
  for (int p=t; p<1024; p+=256){
    float c9[9];
    quant9(xb_[p], xb_[1024+p], xb_[2048+p], wqs, bqs, c9);
    #pragma unroll
    for (int c=0;c<9;++c)
      a9[c][p] = sigmoidf_(S[512+c]*c9[c] + S[768+c]);
  }
  __syncthreads();
  int o = t >> 4, h = t & 15;
  float bias = bc0[o], ls = 0.f, lq = 0.f;
  float* zo = z0 + ((size_t)b*16 + o)*256 + h*16;
  for (int w=0; w<16; ++w){
    float acc = bias;
    for (int ic=0; ic<9; ++ic){
      const float* wk = &wcs[(o*9+ic)*9];
      #pragma unroll
      for (int kh=0; kh<3; ++kh){
        int ih = 2*h-1+kh; if ((unsigned)ih >= 32u) continue;
        #pragma unroll
        for (int kw=0; kw<3; ++kw){
          int iw = 2*w-1+kw; if ((unsigned)iw >= 32u) continue;
          acc += a9[ic][ih*32+iw]*wk[kh*3+kw];
        }
      }
    }
    zo[w] = acc; ls += acc; lq += acc*acc;
  }
  atomicAdd(&ssum[o], ls); atomicAdd(&ssq[o], lq);
  __syncthreads();
  if (t < 16){ atomicAdd(&S[64+t], ssum[t]); atomicAdd(&S[256+64+t], ssq[t]); }
}

// ---------------- K5: bn16+sigmoid -> conv7 (16->32, s2 p1), stats32 ----------------
__global__ void __launch_bounds__(256)
k_conv7(const float* __restrict__ z0,
        const float* __restrict__ wc7, const float* __restrict__ bc7,
        float* __restrict__ S, float* __restrict__ z1){
  __shared__ float a[16][256];
  __shared__ float wcs[32*16*9];
  __shared__ float ssum[32], ssq[32];
  int t = threadIdx.x, b = blockIdx.x;
  if (t < 32){ ssum[t]=0.f; ssq[t]=0.f; }
  for (int i=t; i<4608; i+=256) wcs[i] = wc7[i];
  const float* z = z0 + (size_t)b*4096;
  __syncthreads();
  for (int i=t; i<4096; i+=256){
    int c = i >> 8;
    a[c][i&255] = sigmoidf_(S[512+64+c]*z[i] + S[768+64+c]);
  }
  __syncthreads();
  int o = t >> 3, h = t & 7;
  float bias = bc7[o], ls = 0.f, lq = 0.f;
  float* zo = z1 + ((size_t)b*32 + o)*64 + h*8;
  for (int w=0; w<8; ++w){
    float acc = bias;
    for (int ic=0; ic<16; ++ic){
      const float* wk = &wcs[(o*16+ic)*9];
      #pragma unroll
      for (int kh=0; kh<3; ++kh){
        int ih = 2*h-1+kh; if ((unsigned)ih >= 16u) continue;
        #pragma unroll
        for (int kw=0; kw<3; ++kw){
          int iw = 2*w-1+kw; if ((unsigned)iw >= 16u) continue;
          acc += a[ic][ih*16+iw]*wk[kh*3+kw];
        }
      }
    }
    zo[w] = acc; ls += acc; lq += acc*acc;
  }
  atomicAdd(&ssum[o], ls); atomicAdd(&ssq[o], lq);
  __syncthreads();
  if (t < 32){ atomicAdd(&S[128+t], ssum[t]); atomicAdd(&S[256+128+t], ssq[t]); }
}

// ------- K7: x1 = s6*bn32(z1)+resd(bn16(z0),8); conv13 (32->64 s2 p1); stats64 -------
__global__ void __launch_bounds__(256)
k_res_conv13(const float* __restrict__ z0,
             const float* __restrict__ wc13, const float* __restrict__ bc13,
             const float* __restrict__ w1,
             float* __restrict__ S,
             float* __restrict__ z1 /* in: z1 raw, out: x1 */,
             float* __restrict__ z2){
  __shared__ float a1[32][64];
  __shared__ float ssum[64], ssq[64];
  int t = threadIdx.x, b = blockIdx.x;
  if (t < 64){ ssum[t]=0.f; ssq[t]=0.f; }
  float s6 = sigmoidf_(w1[6]);
  float* z1b = z1 + (size_t)b*2048;
  const float* z0b = z0 + (size_t)b*4096;
  for (int i=t; i<2048; i+=256){
    int c = i>>6, p = i&63, h = p>>3, w = p&7;
    float r = 0.f; int rc = c - 8;
    if ((unsigned)rc < 16u){                       // channel-padded avgpool residual
      float sc = S[512+64+rc], bi = S[768+64+rc];
      const float* zz = z0b + rc*256;
      int y = 2*h, xw = 2*w;
      r = 0.25f*((sc*zz[y*16+xw]+bi) + (sc*zz[y*16+xw+1]+bi)
               + (sc*zz[(y+1)*16+xw]+bi) + (sc*zz[(y+1)*16+xw+1]+bi));
    }
    float xv = s6*(S[512+128+c]*z1b[i] + S[768+128+c]) + r;
    z1b[i] = xv;                                   // store x1 in place
    a1[c][p] = sigmoidf_(xv);
  }
  __syncthreads();
  int o = t >> 2, h = t & 3;
  float bias = bc13[o], ls = 0.f, lq = 0.f;
  float* zo = z2 + ((size_t)b*64 + o)*16 + h*4;
  for (int w=0; w<4; ++w){
    float acc = bias;
    for (int ic=0; ic<32; ++ic){
      const float* wk = wc13 + (o*32+ic)*9;        // L2-resident (73 KB)
      #pragma unroll
      for (int kh=0; kh<3; ++kh){
        int ih = 2*h-1+kh; if ((unsigned)ih >= 8u) continue;
        #pragma unroll
        for (int kw=0; kw<3; ++kw){
          int iw = 2*w-1+kw; if ((unsigned)iw >= 8u) continue;
          acc += a1[ic][ih*8+iw]*wk[kh*3+kw];
        }
      }
    }
    zo[w] = acc; ls += acc; lq += acc*acc;
  }
  atomicAdd(&ssum[o], ls); atomicAdd(&ssq[o], lq);
  __syncthreads();
  if (t < 64){ atomicAdd(&S[192+t], ssum[t]); atomicAdd(&S[256+192+t], ssq[t]); }
}

// -------- K9: x2 = s12*bn64(z2)+resd(x1,16); feat = sigmoid(flatten) fp16 --------
__global__ void k_feat(const float* __restrict__ x1,
                       const float* __restrict__ z2,
                       const float* __restrict__ w1,
                       const float* __restrict__ S,
                       _Float16* __restrict__ feat){
  int t = threadIdx.x, b = blockIdx.x;
  float s12 = sigmoidf_(w1[12]);
  const float* x1b = x1 + (size_t)b*2048;
  const float* z2b = z2 + (size_t)b*1024;
  for (int i=t; i<1024; i+=256){
    int c = i>>4, p = i&15, h = p>>2, w = p&3;
    float r = 0.f; int rc = c - 16;
    if ((unsigned)rc < 32u){
      const float* xx = x1b + rc*64;
      int y = 2*h, xw = 2*w;
      r = 0.25f*(xx[y*8+xw] + xx[y*8+xw+1] + xx[(y+1)*8+xw] + xx[(y+1)*8+xw+1]);
    }
    float xv = s12*(S[512+192+c]*z2b[i] + S[768+192+c]) + r;
    feat[(size_t)b*1024 + i] = (_Float16)sigmoidf_(xv);
  }
}

// ---------------- WMMA GEMM: Out[M,N] = A[M,K] * B[N,K]^T + bias ----------------
// One wave computes a 64x16 output slab: 4 M-tiles reuse one B (weight) fragment
// per K-step -> 5 x 32B-per-lane loads feed 4 v_wmma_f32_16x16x32_f16.
union FragH { v16h v; uint4 q[2]; };

template<int STORE_HALF>
__global__ void __launch_bounds__(256)
k_gemm_wmma(const _Float16* __restrict__ A, int lda,
            const _Float16* __restrict__ B, int ldb,
            const float* __restrict__ bias,
            void* __restrict__ Out, int ldo,
            int Mslabs, int Ntiles, int K){
  int wid  = blockIdx.x*(blockDim.x >> 5) + (threadIdx.x >> 5);
  if (wid >= Mslabs*Ntiles) return;                 // wave-uniform exit
  int lane = threadIdx.x & 31;
  int mS = wid % Mslabs, nT = wid / Mslabs;         // consecutive waves share B rows
  int r  = lane & 15;
  int hi = lane >> 4;
  int m0 = mS*64;
  // A frag (ISA 7.12.2, 16-bit A 16x32): lanes 0-15 K{0..7,16..23}, lanes 16-31 K{8..15,24..31}
  const _Float16* pa = A + (size_t)(m0 + r)*lda + (hi<<3);
  const size_t rowStep = (size_t)16*lda;            // next M-tile within the slab
  // B frag (16-bit B 32x16): lanes 0-15 col N K{0..7,16..23}; lanes 16-31 col N K{8..15,24..31}
  const _Float16* pb = B + (size_t)(nT*16 + r)*ldb + (hi<<3);
  v8f c0 = {}, c1 = {}, c2 = {}, c3 = {};
  for (int k = 0; k < K; k += 32){
    __builtin_prefetch(pb + k + 256, 0, 0);         // global_prefetch_b8, speculative
    __builtin_prefetch(pa + k + 256, 0, 0);
    FragH bfr;
    bfr.q[0] = *(const uint4*)(pb + k);
    bfr.q[1] = *(const uint4*)(pb + k + 16);
    FragH a0, a1, a2, a3;
    a0.q[0] = *(const uint4*)(pa + k);
    a0.q[1] = *(const uint4*)(pa + k + 16);
    a1.q[0] = *(const uint4*)(pa + rowStep + k);
    a1.q[1] = *(const uint4*)(pa + rowStep + k + 16);
    a2.q[0] = *(const uint4*)(pa + 2*rowStep + k);
    a2.q[1] = *(const uint4*)(pa + 2*rowStep + k + 16);
    a3.q[0] = *(const uint4*)(pa + 3*rowStep + k);
    a3.q[1] = *(const uint4*)(pa + 3*rowStep + k + 16);
    c0 = __builtin_amdgcn_wmma_f32_16x16x32_f16(false, a0.v, false, bfr.v, (short)0, c0, false, false);
    c1 = __builtin_amdgcn_wmma_f32_16x16x32_f16(false, a1.v, false, bfr.v, (short)0, c1, false, false);
    c2 = __builtin_amdgcn_wmma_f32_16x16x32_f16(false, a2.v, false, bfr.v, (short)0, c2, false, false);
    c3 = __builtin_amdgcn_wmma_f32_16x16x32_f16(false, a3.v, false, bfr.v, (short)0, c3, false, false);
  }
  int nn = nT*16 + r;
  float bv = bias[nn];
  // C/D layout: lane -> col (lane&15); VGPR i -> row i + 8*(lane>>4)
  v8f acc[4] = {c0, c1, c2, c3};
  #pragma unroll
  for (int tM = 0; tM < 4; ++tM){
    int mbase = m0 + tM*16 + (hi<<3);
    #pragma unroll
    for (int i=0; i<8; ++i){
      float val = acc[tM][i] + bv;
      size_t idx = (size_t)(mbase + i)*ldo + nn;
      if (STORE_HALF) ((_Float16*)Out)[idx] = (_Float16)val;
      else            ((float*)Out)[idx]    = val;
    }
  }
}

// ---------------- group-sum reduce: [2048,2000] -> [2048,10] ----------------
__global__ void k_reduce(const float* __restrict__ out2, float* __restrict__ out){
  int i = blockIdx.x*blockDim.x + threadIdx.x;
  if (i >= 2048*10) return;
  int b = i/10, g = i%10;
  const float* p = out2 + (size_t)b*2000 + g*200;
  float s = 0.f;
  for (int j=0; j<200; ++j) s += p[j];
  out[i] = s*0.1f;
}

extern "C" void kernel_launch(void* const* d_in, const int* in_sizes, int n_in,
                              void* d_out, int out_size, void* d_ws, size_t ws_size,
                              hipStream_t stream){
  const float* x    = (const float*)d_in[0];
  const float* wqr  = (const float*)d_in[1];
  const float* bqr  = (const float*)d_in[2];
  const float* wqg  = (const float*)d_in[3];
  const float* bqg  = (const float*)d_in[4];
  const float* wqb  = (const float*)d_in[5];
  const float* bqb  = (const float*)d_in[6];
  const float* wc0  = (const float*)d_in[7];
  const float* bc0  = (const float*)d_in[8];
  const float* wc7  = (const float*)d_in[9];
  const float* bc7  = (const float*)d_in[10];
  const float* wc13 = (const float*)d_in[11];
  const float* bc13 = (const float*)d_in[12];
  const float* wfc3 = (const float*)d_in[13];
  const float* bfc3 = (const float*)d_in[14];
  const float* wfc4 = (const float*)d_in[15];
  const float* bfc4 = (const float*)d_in[16];
  const float* w1   = (const float*)d_in[17];

  char* ws = (char*)d_ws;
  // time-overlapped layout (~71 MB):
  // region0 [0, 33.55MB): z0 (K3..K7); feat@0 (K9..K10); out2@8,388,608 (K11..K12)
  // region1 [33.55MB, 50.33MB): z1/x1 (K5..K9); A2 (K10..K11)
  float*    z0   = (float*)   (ws + 0);
  _Float16* feat = (_Float16*)(ws + 0);
  float*    out2 = (float*)   (ws + 8388608);
  float*    z1   = (float*)   (ws + 33554432);
  _Float16* A2   = (_Float16*)(ws + 33554432);
  float*    z2   = (float*)   (ws + 50331648);
  _Float16* w3h  = (_Float16*)(ws + 58720256);   // [2000,1024]
  _Float16* w4h  = (_Float16*)(ws + 62816256);   // [2000,2048] zero-padded K
  float*    S    = (float*)   (ws + 71008256);   // 1024 floats of BN state

  k_zero_u32<<<2, 256, 0, stream>>>((uint32_t*)S, 512);
  k_cvt_pad<<<(2000*1024+255)/256, 256, 0, stream>>>(wfc3, w3h, 2000, 1024, 1024);
  k_cvt_pad<<<(2000*2048+255)/256, 256, 0, stream>>>(wfc4, w4h, 2000, 2000, 2048);

  k_stats9<<<2048, 256, 0, stream>>>(x, wqr,bqr,wqg,bqg,wqb,bqb, S);
  k_finalize<<<1, 64, 0, stream>>>(S, 0, 9,  1.f/(2048.f*1024.f));

  k_conv0<<<2048, 256, 0, stream>>>(x, wqr,bqr,wqg,bqg,wqb,bqb, wc0, bc0, S, z0);
  k_finalize<<<1, 64, 0, stream>>>(S, 1, 16, 1.f/(2048.f*256.f));

  k_conv7<<<2048, 256, 0, stream>>>(z0, wc7, bc7, S, z1);
  k_finalize<<<1, 64, 0, stream>>>(S, 2, 32, 1.f/(2048.f*64.f));

  k_res_conv13<<<2048, 256, 0, stream>>>(z0, wc13, bc13, w1, S, z1, z2);
  k_finalize<<<1, 64, 0, stream>>>(S, 3, 64, 1.f/(2048.f*16.f));

  k_feat<<<2048, 256, 0, stream>>>(z1, z2, w1, S, feat);

  // zero A2 (covers fp16 pad columns 2000..2047) after x1's last use
  k_zero_u32<<<(2097152+255)/256, 256, 0, stream>>>((uint32_t*)A2, 2097152);

  // fc3: [2048,1024] x [2000,1024]^T -> A2 fp16 [2048,2048-padded]
  int waves = 32*125;                  // 4000 waves of 64x16 slabs, 8 waves/block
  k_gemm_wmma<1><<<(waves+7)/8, 256, 0, stream>>>(feat, 1024, w3h, 1024, bfc3,
                                                  (void*)A2, 2048, 32, 125, 1024);
  // fc4: [2048,2048] x [2000,2048]^T -> out2 f32 [2048,2000]
  k_gemm_wmma<0><<<(waves+7)/8, 256, 0, stream>>>(A2, 2048, w4h, 2048, bfc4,
                                                  (void*)out2, 2000, 32, 125, 2048);

  k_reduce<<<(20480+255)/256, 256, 0, stream>>>(out2, (float*)d_out);
}
